// AttnBlock_7060926234866
// MI455X (gfx1250) — matrix operand order repaired and checked by reading, not compiled, not run on hardware
//
#include <hip/hip_runtime.h>
#include <hip/hip_bf16.h>
#include <cmath>

typedef __attribute__((ext_vector_type(16))) _Float16 v16h;
typedef __attribute__((ext_vector_type(8)))  _Float16 v8h;
typedef __attribute__((ext_vector_type(8)))  float    v8f;

#define C_DIM 512
#define N_TOK 4096
#define B_SZ  4
#define M_TOT (B_SZ * N_TOK)   // 16384
#define NGRP  32
#define GCH   (C_DIM / NGRP)   // 16

// ---------------- WMMA fragment helpers (wave32, gfx1250 layouts) ----------------

__device__ __forceinline__ v16h frag_cat(v8h lo, v8h hi) {
  v16h r;
#pragma unroll
  for (int i = 0; i < 8; ++i) { r[i] = lo[i]; r[8 + i] = hi[i]; }
  return r;
}

// A fragment 16x32 f16 from row-major src (leading dim ld elements).
// Lane: row = m0 + lane%16; half h=lane/16: elems 0..7 -> k0+8h+i, 8..15 -> k0+16+8h+i.
__device__ __forceinline__ v16h load_frag_a(const _Float16* src, int ld, int m0, int k0) {
  const int lane = threadIdx.x & 31;
  const int h = lane >> 4;
  const _Float16* p = src + (size_t)(m0 + (lane & 15)) * ld + k0 + 8 * h;
  v8h lo = *(const v8h*)p;
  v8h hi = *(const v8h*)(p + 16);
  return frag_cat(lo, hi);
}

// B fragment 32x16 f16 where B[k][n] = srcT[n][k], srcT row-major (ld elements).
// Lane: col n = n0 + lane%16; half h=lane/16: elems 0..7 -> k0+8h+i, 8..15 -> k0+16+8h+i.
__device__ __forceinline__ v16h load_frag_b(const _Float16* srcT, int ld, int n0, int k0) {
  const int lane = threadIdx.x & 31;
  const int h = lane >> 4;
  const _Float16* p = srcT + (size_t)(n0 + (lane & 15)) * ld + k0 + 8 * h;
  v8h lo = *(const v8h*)p;
  v8h hi = *(const v8h*)(p + 16);
  return frag_cat(lo, hi);
}

__device__ __forceinline__ v8f wmma_f16(v16h a, v16h b, v8f c) {
  return __builtin_amdgcn_wmma_f32_16x16x32_f16(false, a, false, b, (short)0, c, false, false);
}

// Async global->LDS 16-byte copy (CDNA5 direct path, tracked by ASYNCcnt).
// LDS byte address = low 32 bits of the generic pointer (flat->LDS maps addr[31:0]).
__device__ __forceinline__ void async_copy16(void* lds_ptr, const void* gptr) {
  uint32_t lds_off = (uint32_t)(uintptr_t)lds_ptr;
  uint64_t gaddr   = (uint64_t)(uintptr_t)gptr;
  asm volatile("global_load_async_to_lds_b128 %0, %1, off"
               :: "v"(lds_off), "v"(gaddr)
               : "memory");
}

__device__ __forceinline__ void wait_async_all() {
  asm volatile("s_wait_asynccnt 0x0" ::: "memory");
}

// ---------------- 1) GroupNorm statistics ----------------
// One 256-thread block per (batch, group); reduce 4096*16 elements.
__global__ __launch_bounds__(256) void gn_stats_kernel(const float* __restrict__ x,
                                                       float* __restrict__ stats) {
  const int bg = blockIdx.x;            // 0..127
  const int b = bg >> 5, g = bg & 31;
  const int tid = threadIdx.x;
  float s = 0.f, s2 = 0.f;
  for (int i = tid; i < N_TOK * GCH; i += 256) {
    int hw = i >> 4, cc = i & 15;
    float v = x[((size_t)(b * N_TOK + hw)) * C_DIM + g * GCH + cc];
    s += v; s2 += v * v;
  }
  __shared__ float rs[256], rq[256];
  rs[tid] = s; rq[tid] = s2;
  __syncthreads();
  for (int off = 128; off > 0; off >>= 1) {
    if (tid < off) { rs[tid] += rs[tid + off]; rq[tid] += rq[tid + off]; }
    __syncthreads();
  }
  if (tid == 0) {
    float inv_n = 1.0f / (float)(N_TOK * GCH);
    float mean = rs[0] * inv_n;
    float var = rq[0] * inv_n - mean * mean;
    stats[bg * 2 + 0] = mean;
    stats[bg * 2 + 1] = rsqrtf(var + 1e-6f);
  }
}

// ---------------- 2) Normalize + affine, convert to f16 ----------------
__global__ __launch_bounds__(256) void gn_apply_kernel(const float* __restrict__ x,
                                                       const float* __restrict__ stats,
                                                       const float* __restrict__ scale,
                                                       const float* __restrict__ bias,
                                                       _Float16* __restrict__ xo) {
  size_t idx = (size_t)blockIdx.x * blockDim.x + threadIdx.x;
  if (idx >= (size_t)M_TOT * C_DIM) return;
  int c = (int)(idx & (C_DIM - 1));
  int bn = (int)(idx >> 9);
  int b = bn >> 12;
  int g = c >> 4;
  float mean = stats[(b * NGRP + g) * 2 + 0];
  float rstd = stats[(b * NGRP + g) * 2 + 1];
  xo[idx] = (_Float16)((x[idx] - mean) * rstd * scale[c] + bias[c]);
}

// ---------------- 3) Weight transpose + f16 convert: Wt[n][k] = W[k][n] ----------------
__global__ __launch_bounds__(256) void wtrans_kernel(const float* __restrict__ W,
                                                     _Float16* __restrict__ Wt) {
  int idx = blockIdx.x * blockDim.x + threadIdx.x;   // 512*512
  int k = idx >> 9, n = idx & (C_DIM - 1);
  Wt[(size_t)n * C_DIM + k] = (_Float16)W[(size_t)k * C_DIM + n];
}

// ---------------- 4/6) WMMA GEMM: C[m,n] = A[m,:]·Wt[n,:] + bias[n] ----------------
// mode 0: out16[m*512+n] = (acc+bias)*scl          (Q*s / K*s)
// mode 1: out16[(b*512+n)*4096 + r] = acc+bias     (V stored transposed per batch)
// mode 2: out32[m*512+n] = acc+bias + residual     (final projection, fp32)
__global__ __launch_bounds__(128) void gemm_kernel(const _Float16* __restrict__ A,
                                                   const _Float16* __restrict__ Wt,
                                                   const float* __restrict__ bias,
                                                   _Float16* __restrict__ out16,
                                                   float* __restrict__ out32,
                                                   const float* __restrict__ residual,
                                                   float scl, int mode) {
  const int wid = threadIdx.x >> 5;
  const int m0 = (blockIdx.x * 4 + wid) * 16;   // m-tile (16 rows)
  const int n0 = blockIdx.y * 64;               // 4 n-tiles of 16
  v8f acc[4];
#pragma unroll
  for (int t = 0; t < 4; ++t) acc[t] = (v8f){0.f,0.f,0.f,0.f,0.f,0.f,0.f,0.f};

#pragma unroll 4
  for (int k0 = 0; k0 < C_DIM; k0 += 32) {
    v16h a = load_frag_a(A, C_DIM, m0, k0);
#pragma unroll
    for (int t = 0; t < 4; ++t) {
      v16h bf = load_frag_b(Wt, C_DIM, n0 + t * 16, k0);
      acc[t] = wmma_f16(a, bf, acc[t]);
    }
  }

  const int lane = threadIdx.x & 31;
  const int nl = lane & 15, h = lane >> 4;
#pragma unroll
  for (int t = 0; t < 4; ++t) {
    int n = n0 + t * 16 + nl;
    float bn = bias[n];
#pragma unroll
    for (int j = 0; j < 8; ++j) {
      int m = m0 + j + 8 * h;
      float v = acc[t][j] + bn;
      if (mode == 0) {
        out16[(size_t)m * C_DIM + n] = (_Float16)(v * scl);
      } else if (mode == 1) {
        int b = m >> 12, r = m & (N_TOK - 1);
        out16[((size_t)(b * C_DIM + n)) * N_TOK + r] = (_Float16)v;
      } else {
        size_t o = (size_t)m * C_DIM + n;
        out32[o] = v + residual[o];
      }
    }
  }
}

// ---------------- 5) Flash attention (online softmax), WMMA + async LDS staging ----
// Block: 128 threads = 4 waves; each wave owns 16 query rows of a 64-row tile.
// Loop over 32-key blocks: S = Q·K^T (2 frags), online softmax, O += P·V (32 frags).
// K / V^T / Q tiles are staged with global_load_async_to_lds_b128 (ASYNCcnt path),
// next block is prefetched into cache during compute.
#define BM 64
#define BN 32
__global__ __launch_bounds__(128) void flash_attn_kernel(const _Float16* __restrict__ Q,
                                                         const _Float16* __restrict__ K,
                                                         const _Float16* __restrict__ VT,
                                                         _Float16* __restrict__ O16) {
  __shared__ __align__(16) _Float16 sQ[BM * C_DIM];    // 64 KB
  __shared__ __align__(16) _Float16 sK[BN * C_DIM];    // 32 KB
  __shared__ __align__(16) _Float16 sVT[C_DIM * BN];   // 32 KB
  __shared__ __align__(16) _Float16 sP[BM * BN];       //  4 KB

  const int b = blockIdx.y;
  const int qt = blockIdx.x;
  const int tid = threadIdx.x;
  const int wid = tid >> 5, lane = tid & 31;
  const int nl = lane & 15, h = lane >> 4;
  const int m0 = wid * 16;

  // stage Q tile (async; consumed after the first wait+barrier inside the loop)
  const _Float16* Qg = Q + (size_t)(b * N_TOK + qt * BM) * C_DIM;
#pragma unroll 4
  for (int i = tid; i < (BM * C_DIM) / 8; i += 128)
    async_copy16(&sQ[i * 8], Qg + (size_t)i * 8);

  float mrow[8], lrow[8];
#pragma unroll
  for (int j = 0; j < 8; ++j) { mrow[j] = -1e30f; lrow[j] = 0.f; }
  v8f o[32];
#pragma unroll
  for (int c = 0; c < 32; ++c) o[c] = (v8f){0.f,0.f,0.f,0.f,0.f,0.f,0.f,0.f};

  const _Float16* Vg = VT + (size_t)b * C_DIM * N_TOK;

  for (int t = 0; t < N_TOK / BN; ++t) {
    __syncthreads();   // previous sK/sVT consumers done
    const _Float16* Kg = K + (size_t)(b * N_TOK + t * BN) * C_DIM;
#pragma unroll 4
    for (int i = tid; i < (BN * C_DIM) / 8; i += 128)
      async_copy16(&sK[i * 8], Kg + (size_t)i * 8);
#pragma unroll 4
    for (int i = tid; i < (C_DIM * BN) / 8; i += 128) {
      int c = i >> 2, rv = i & 3;   // 4 vec8 per channel row
      async_copy16(&sVT[i * 8], Vg + (size_t)c * N_TOK + t * BN + rv * 8);
    }
    // prefetch next key block into cache while this one is consumed
    if (t + 1 < N_TOK / BN) {
      const _Float16* Kn = K + (size_t)(b * N_TOK + (t + 1) * BN) * C_DIM;
      __builtin_prefetch(Kn + (size_t)tid * 128, 0, 0);
      __builtin_prefetch(Vg + (size_t)(tid * 4) * N_TOK + (t + 1) * BN, 0, 0);
    }
    wait_async_all();  // my async LDS writes complete
    __syncthreads();   // everyone's complete -> tiles visible

    // S = Q K^T for this wave's 16 rows (two 16x16 frags over BN=32 keys)
    v8f s0 = (v8f){0.f,0.f,0.f,0.f,0.f,0.f,0.f,0.f};
    v8f s1 = (v8f){0.f,0.f,0.f,0.f,0.f,0.f,0.f,0.f};
#pragma unroll 4
    for (int k0 = 0; k0 < C_DIM; k0 += 32) {
      v16h a  = load_frag_a(sQ, C_DIM, m0, k0);
      v16h b0 = load_frag_b(sK, C_DIM, 0,  k0);
      v16h b1 = load_frag_b(sK, C_DIM, 16, k0);
      s0 = wmma_f16(a, b0, s0);
      s1 = wmma_f16(a, b1, s1);
    }

    // online softmax; C-layout: VGPR j holds row m0+j+8h, cols across 16-lane half
#pragma unroll
    for (int j = 0; j < 8; ++j) {
      float mx = fmaxf(s0[j], s1[j]);
#pragma unroll
      for (int d = 8; d >= 1; d >>= 1) mx = fmaxf(mx, __shfl_xor(mx, d, 16));
      float mnew = fmaxf(mrow[j], mx);
      float corr = __expf(mrow[j] - mnew);
      float e0 = __expf(s0[j] - mnew);
      float e1 = __expf(s1[j] - mnew);
      float rs = e0 + e1;
#pragma unroll
      for (int d = 8; d >= 1; d >>= 1) rs += __shfl_xor(rs, d, 16);
      lrow[j] = lrow[j] * corr + rs;
      mrow[j] = mnew;
      s0[j] = e0; s1[j] = e1;
#pragma unroll
      for (int c = 0; c < 32; ++c) o[c][j] *= corr;
    }

    // spill P (f16) to LDS in row-major for A-fragment reload (wave-private rows)
#pragma unroll
    for (int j = 0; j < 8; ++j) {
      int mr = m0 + j + 8 * h;
      sP[mr * BN + nl]      = (_Float16)s0[j];
      sP[mr * BN + 16 + nl] = (_Float16)s1[j];
    }
    asm volatile("s_wait_dscnt 0x0" ::: "memory");   // LDS RAW fence (same wave)

    // O += P · V  (A = P 16x32; B frags from V^T, contiguous per lane)
    v16h pa = load_frag_a(sP, BN, m0, 0);
#pragma unroll
    for (int c = 0; c < 32; ++c) {
      v16h bf = load_frag_b(sVT, BN, c * 16, 0);
      o[c] = wmma_f16(pa, bf, o[c]);
    }
  }

  // epilogue: normalize by l, store f16
  _Float16* Og = O16 + (size_t)(b * N_TOK + qt * BM) * C_DIM;
#pragma unroll
  for (int j = 0; j < 8; ++j) {
    float inv = 1.0f / lrow[j];
    int m = m0 + j + 8 * h;
#pragma unroll
    for (int c = 0; c < 32; ++c)
      Og[(size_t)m * C_DIM + c * 16 + nl] = (_Float16)(o[c][j] * inv);
  }
}

// ---------------- host orchestration ----------------
extern "C" void kernel_launch(void* const* d_in, const int* in_sizes, int n_in,
                              void* d_out, int out_size, void* d_ws, size_t ws_size,
                              hipStream_t stream) {
  const float* x        = (const float*)d_in[0];
  const float* gn_scale = (const float*)d_in[1];
  const float* gn_bias  = (const float*)d_in[2];
  const float* Wq = (const float*)d_in[3];  const float* bq = (const float*)d_in[4];
  const float* Wk = (const float*)d_in[5];  const float* bk = (const float*)d_in[6];
  const float* Wv = (const float*)d_in[7];  const float* bv = (const float*)d_in[8];
  const float* Wp = (const float*)d_in[9];  const float* bp = (const float*)d_in[10];
  float* out = (float*)d_out;

  size_t off = 0;
  auto alloc = [&](size_t bytes) -> void* {
    void* p = (char*)d_ws + off;
    off += (bytes + 255) & ~(size_t)255;
    return p;
  };
  float*     stats = (float*)alloc(B_SZ * NGRP * 2 * sizeof(float));
  _Float16*  x16   = (_Float16*)alloc((size_t)M_TOT * C_DIM * 2);
  _Float16*  Wqt   = (_Float16*)alloc((size_t)C_DIM * C_DIM * 2);
  _Float16*  Wkt   = (_Float16*)alloc((size_t)C_DIM * C_DIM * 2);
  _Float16*  Wvt   = (_Float16*)alloc((size_t)C_DIM * C_DIM * 2);
  _Float16*  Wpt   = (_Float16*)alloc((size_t)C_DIM * C_DIM * 2);
  _Float16*  q16   = (_Float16*)alloc((size_t)M_TOT * C_DIM * 2);
  _Float16*  k16   = (_Float16*)alloc((size_t)M_TOT * C_DIM * 2);
  _Float16*  vt16  = (_Float16*)alloc((size_t)M_TOT * C_DIM * 2);
  _Float16*  o16   = x16;   // reuse: x16 dead after QKV GEMMs

  const float qk_scale = 1.0f / sqrtf(sqrtf((float)C_DIM));

  // 1) group-norm stats
  gn_stats_kernel<<<B_SZ * NGRP, 256, 0, stream>>>(x, stats);
  // 2) normalize -> f16
  {
    size_t total = (size_t)M_TOT * C_DIM;
    gn_apply_kernel<<<(int)((total + 255) / 256), 256, 0, stream>>>(x, stats, gn_scale, gn_bias, x16);
  }
  // 3) weights -> f16 transposed
  wtrans_kernel<<<(C_DIM * C_DIM) / 256, 256, 0, stream>>>(Wq, Wqt);
  wtrans_kernel<<<(C_DIM * C_DIM) / 256, 256, 0, stream>>>(Wk, Wkt);
  wtrans_kernel<<<(C_DIM * C_DIM) / 256, 256, 0, stream>>>(Wv, Wvt);
  wtrans_kernel<<<(C_DIM * C_DIM) / 256, 256, 0, stream>>>(Wp, Wpt);
  // 4) Q, K (pre-scaled), V (stored transposed)
  dim3 ggrid(M_TOT / 64, C_DIM / 64);
  gemm_kernel<<<ggrid, 128, 0, stream>>>(x16, Wqt, bq, q16, nullptr, nullptr, qk_scale, 0);
  gemm_kernel<<<ggrid, 128, 0, stream>>>(x16, Wkt, bk, k16, nullptr, nullptr, qk_scale, 0);
  gemm_kernel<<<ggrid, 128, 0, stream>>>(x16, Wvt, bv, vt16, nullptr, nullptr, 1.0f, 1);
  // 5) flash attention
  flash_attn_kernel<<<dim3(N_TOK / BM, B_SZ), 128, 0, stream>>>(q16, k16, vt16, o16);
  // 6) projection + bias + residual -> fp32 output
  gemm_kernel<<<ggrid, 128, 0, stream>>>(o16, Wpt, bp, nullptr, out, x, 1.0f, 2);
}
